// Codebook_26259430048749
// MI455X (gfx1250) — matrix operand order repaired and checked by reading, not compiled, not run on hardware
//
#include <hip/hip_runtime.h>
#include <hip/hip_bf16.h>

typedef __attribute__((ext_vector_type(16))) __bf16 v16bf;
typedef __attribute__((ext_vector_type(8)))  __bf16 v8bf;
typedef __attribute__((ext_vector_type(8)))  float  v8f;
typedef __attribute__((ext_vector_type(4)))  unsigned int u32x4;
typedef __attribute__((ext_vector_type(8)))  int          i32x8;
typedef __attribute__((ext_vector_type(4)))  int          i32x4;

#define N_TOT   32768
#define D_DIM   256
#define K_DIM   1024
#define TILE_N  128
#define LDS_STR 264      // 256 + 8 bf16 pad
#define SLAB_E  4096     // one B slab: 256 d-rows x 16 k-cols (bf16 elements)

#if defined(__has_builtin)
#if __has_builtin(__builtin_amdgcn_tensor_load_to_lds)
#define USE_TDM 1
#endif
#endif
#ifndef USE_TDM
#define USE_TDM 0
#endif

// ---------------------------------------------------------------------------
// Prep 1: split emb into bf16 hi/lo, stored TRANSPOSED [D][K].
// ---------------------------------------------------------------------------
__global__ __launch_bounds__(256) void vq_decompose_kernel(
    const float* __restrict__ emb,
    unsigned short* __restrict__ ehiT_u,
    unsigned short* __restrict__ eloT_u)
{
    int t = blockIdx.x * 256 + threadIdx.x;     // 0 .. K*D-1
    int k = t & (K_DIM - 1);
    int d = t >> 10;
    float f = emb[(size_t)k * D_DIM + d];
    __bf16 h = (__bf16)f;
    __bf16 l = (__bf16)(f - (float)h);
    __bf16* ehiT = (__bf16*)ehiT_u;
    __bf16* eloT = (__bf16*)eloT_u;
    ehiT[(size_t)d * K_DIM + k] = h;
    eloT[(size_t)d * K_DIM + k] = l;
}

// ---------------------------------------------------------------------------
// Prep 2: half squared norms; zero the loss cell each call (determinism).
// ---------------------------------------------------------------------------
__global__ __launch_bounds__(256) void vq_norms_kernel(
    const float* __restrict__ emb,
    float* __restrict__ hnorm,
    float* __restrict__ loss)
{
    int k = blockIdx.x * 256 + threadIdx.x;     // 0 .. 1023
    const float* e = emb + (size_t)k * D_DIM;
    float s = 0.f;
    #pragma unroll 8
    for (int d = 0; d < D_DIM; ++d) { float v = e[d]; s += v * v; }
    hnorm[k] = 0.5f * s;
    if (k == 0) *loss = 0.f;
}

// ---------------------------------------------------------------------------
// TDM: DMA one [256 x 16] bf16 tile (rows strided 1024 elems in global)
// into LDS, packed as contiguous 32B rows. D# per CDNA5 ISA §8.3/8.4.
// 6-arg builtin: (u32x4 g0, i32x8 g1, i32x4 g2, i32x4 g3, i32x8, i32 cpol)
// ---------------------------------------------------------------------------
#if USE_TDM
__device__ __forceinline__ void tdm_load_slab(unsigned lds_off, const __bf16* gptr)
{
    unsigned long long ga = (unsigned long long)gptr;
    u32x4 g0 = { 1u,                                   // count=1, user mode
                 lds_off,                              // lds_addr [63:32]
                 (unsigned)ga,                         // global_addr lo [95:64]
                 (unsigned)(ga >> 32) | (2u << 30) };  // addr[56:32] | type=2
    i32x8 g1 = { 0x00010000,                 // data_size=1 (2B), no pad/iter/mcast
                 (int)(1024u << 16),         // tensor_dim0 = 1024     [79:48]
                 (int)(256u << 16),          // tensor_dim1 = 256      [111:80]
                 (int)(16u  << 16),          // tile_dim0   = 16       [127:112]
                 256,                        // tile_dim1   = 256      [143:128]
                 1024,                       // tensor_dim0_stride     [207:160]
                 0, 0 };
    i32x4 z4 = { 0, 0, 0, 0 };
    i32x8 z8 = { 0, 0, 0, 0, 0, 0, 0, 0 };
    __builtin_amdgcn_tensor_load_to_lds(g0, g1, z4, z4, z8, 0);
}
#endif

// Stage the hi/lo B slabs for k-tile kt into LDS at element offset dstoff.
__device__ __forceinline__ void stage_slab(int kt, __bf16* sbase, int dstoff,
                                           const __bf16* ehiT, const __bf16* eloT,
                                           int wave, int t)
{
#if USE_TDM
    if (wave == 0) {
        tdm_load_slab((unsigned)(unsigned long long)(sbase + dstoff),
                      ehiT + kt * 16);
        tdm_load_slab((unsigned)(unsigned long long)(sbase + dstoff + SLAB_E),
                      eloT + kt * 16);
    }
#else
    // cooperative fallback: 256 threads copy one 32B row each
    v16bf h = *(const v16bf*)(ehiT + (size_t)t * K_DIM + kt * 16);
    v16bf l = *(const v16bf*)(eloT + (size_t)t * K_DIM + kt * 16);
    *(v16bf*)(sbase + dstoff + t * 16)          = h;
    *(v16bf*)(sbase + dstoff + SLAB_E + t * 16) = l;
#endif
}

__device__ __forceinline__ void stage_wait()
{
#if USE_TDM
    __builtin_amdgcn_s_wait_tensorcnt((short)0);   // no-op for non-issuing waves
#endif
    __syncthreads();
}

// ---------------------------------------------------------------------------
// Main: split-bf16 WMMA GEMM (z.e^T - 0.5||e||^2) + fused row argmax.
// 256 threads = 8 waves; tile = 128 z-rows x all 1024 codes.
// LDS: z tile (132KB) consumed once into VGPRs, then ALIASED as a
// double-buffered TDM B-slab arena (4 x 8KB), addressed with integer
// offsets from ONE addrspace(3) base ; B operands are gathered one bf16 at a time.
// ---------------------------------------------------------------------------
__global__ __launch_bounds__(256) void vq_argmin_kernel(
    const float* __restrict__ x,                 // [32, 256, 1024]
    const unsigned short* __restrict__ ehiT_u,   // [D][K] bf16
    const unsigned short* __restrict__ eloT_u,   // [D][K] bf16
    const float* __restrict__ hnorm,             // [K]
    int* __restrict__ idx_out)                   // [N]
{
    __shared__ __align__(16) __bf16 smem[2 * TILE_N * LDS_STR];
    __bf16* sbase = smem;
    __bf16* zhi   = smem;
    __bf16* zlo   = smem + TILE_N * LDS_STR;

    const int tile = blockIdx.x;                 // 0..255
    const int t    = threadIdx.x;
    const int b    = tile >> 3;
    const int hwb  = (tile & 7) * TILE_N;

    // ---- stage split-bf16 z tile into LDS (coalesced over hw) ----
    {
        const int n  = t & 127;
        const int dp = t >> 7;
        const float* xb = x + (size_t)b * (D_DIM * 1024) + hwb + n;
        #pragma unroll 4
        for (int it = 0; it < 128; ++it) {
            int d   = it * 2 + dp;
            float f = xb[(size_t)d * 1024];
            __bf16 h = (__bf16)f;
            __bf16 l = (__bf16)(f - (float)h);
            zhi[n * LDS_STR + d] = h;
            zlo[n * LDS_STR + d] = l;
        }
    }
    __syncthreads();

    const int lane = t & 31;
    const int wave = t >> 5;
    const int l16  = lane & 15;
    const int half = lane >> 4;
    const int row  = wave * 16 + l16;

    // ---- hoist full A operand (16 rows x 256 D, hi+lo) into VGPRs ----
    v16bf ahi[8], alo[8];
    #pragma unroll
    for (int ds = 0; ds < 8; ++ds) {
        const __bf16* ph = &zhi[row * LDS_STR + ds * 32 + half * 8];
        v8bf h0 = *(const v8bf*)(ph);
        v8bf h1 = *(const v8bf*)(ph + 16);
        ahi[ds] = __builtin_shufflevector(h0, h1,
                   0,1,2,3,4,5,6,7,8,9,10,11,12,13,14,15);
        const __bf16* pl = &zlo[row * LDS_STR + ds * 32 + half * 8];
        v8bf l0 = *(const v8bf*)(pl);
        v8bf l1 = *(const v8bf*)(pl + 16);
        alo[ds] = __builtin_shufflevector(l0, l1,
                   0,1,2,3,4,5,6,7,8,9,10,11,12,13,14,15);
    }
    __syncthreads();   // z LDS fully consumed -> safe to overwrite with B slabs

    const __bf16* ehiT = (const __bf16*)ehiT_u;
    const __bf16* eloT = (const __bf16*)eloT_u;

    float best_v[8];
    int   best_k[8];
    #pragma unroll
    for (int j = 0; j < 8; ++j) { best_v[j] = -3.0e38f; best_k[j] = 0; }

    // prologue: stage k-tile 0 into buffer 0 (element offset 0)
    stage_slab(0, sbase, 0, ehiT, eloT, wave, t);
    stage_wait();

    const int rr = (lane) * 16;                  // lane's packed 32B row base

    for (int kt = 0; kt < K_DIM / 16; ++kt) {
        const int curoff = (kt & 1) * (2 * SLAB_E);   // buffer select (elements)
        const int nxtoff = curoff ^ (2 * SLAB_E);
        if (kt + 1 < K_DIM / 16)
            stage_slab(kt + 1, sbase, nxtoff, ehiT, eloT, wave, t);

        v8f acc = {};
        #pragma unroll
        for (int ds = 0; ds < 8; ++ds) {
            const int ro = curoff + ds * (32 * 16) + l16;
            v16bf bh, bl;
            #pragma unroll
            for (int s = 0; s < 16; ++s) {
                const int kk = (s < 8) ? (half * 8 + s) : (16 + half * 8 + (s - 8));
                bh[s] = sbase[ro + kk * 16];
                bl[s] = sbase[ro + kk * 16 + SLAB_E];
            }
            acc = __builtin_amdgcn_wmma_f32_16x16x32_bf16(
                      false, ahi[ds], false, bh, (short)0, acc, false, false);
            acc = __builtin_amdgcn_wmma_f32_16x16x32_bf16(
                      false, ahi[ds], false, bl, (short)0, acc, false, false);
            acc = __builtin_amdgcn_wmma_f32_16x16x32_bf16(
                      false, alo[ds], false, bh, (short)0, acc, false, false);
        }

        const int kc = kt * 16;
        float hn = hnorm[kc + l16];
        #pragma unroll
        for (int j = 0; j < 8; ++j) {
            float v = acc[j] - hn;                  // z.e - 0.5||e||^2
            if (v > best_v[j]) { best_v[j] = v; best_k[j] = kc + l16; }
        }

        stage_wait();   // kt+1 DMA complete + all waves done with cur buffer
    }

    // ---- argmax reduce across the 16 lanes holding columns 0..15 ----
    #pragma unroll
    for (int j = 0; j < 8; ++j) {
        float v = best_v[j];
        int   k = best_k[j];
        #pragma unroll
        for (int m = 8; m >= 1; m >>= 1) {
            float ov = __shfl_xor(v, m, 32);
            int   ok = __shfl_xor(k, m, 32);
            if (ov > v || (ov == v && ok < k)) { v = ov; k = ok; }
        }
        if (l16 == 0) {
            int mrow = j + half * 8;
            idx_out[tile * TILE_N + wave * 16 + mrow] = k;
        }
    }
}

// ---------------------------------------------------------------------------
// Gather emb[idx] back to NCHW (coalesced over hw) + fused loss.
// ---------------------------------------------------------------------------
__global__ __launch_bounds__(256) void vq_gather_loss_kernel(
    const float* __restrict__ x,
    const float* __restrict__ emb,
    const int* __restrict__ idx,
    float* __restrict__ out,
    float* __restrict__ loss)
{
    const int bd = blockIdx.x;                   // 0 .. 32*256-1
    const int b  = bd >> 8;
    const int d  = bd & 255;
    const int t  = threadIdx.x;
    const size_t base = (size_t)b * (D_DIM * 1024) + (size_t)d * 1024;

    float local = 0.f;
    #pragma unroll
    for (int i = 0; i < 4; ++i) {
        int hw = i * 256 + t;
        int n  = b * 1024 + hw;
        int id = idx[n];
        float q  = emb[(size_t)id * D_DIM + d];
        float xv = x[base + hw];
        out[base + hw] = q;
        float df = q - xv;
        local += df * df;
    }

    __shared__ float red[256];
    red[t] = local;
    __syncthreads();
    for (int s = 128; s > 0; s >>= 1) {
        if (t < s) red[t] += red[t + s];
        __syncthreads();
    }
    if (t == 0)
        atomicAdd(loss, red[0] * (1.25f / 8388608.0f));
}

// ---------------------------------------------------------------------------
extern "C" void kernel_launch(void* const* d_in, const int* in_sizes, int n_in,
                              void* d_out, int out_size, void* d_ws, size_t ws_size,
                              hipStream_t stream)
{
    const float* x   = (const float*)d_in[0];   // [32,256,32,32] fp32
    const float* emb = (const float*)d_in[1];   // [1024,256]     fp32

    float* out  = (float*)d_out;                            // z_quant (8388608)
    int*   idxp = (int*)((float*)d_out + 8388608);          // idx (32768)
    float* loss = (float*)d_out + 8388608 + 32768;          // scalar loss

    char* ws = (char*)d_ws;
    unsigned short* ehiT = (unsigned short*)(ws);                 // 512 KB
    unsigned short* eloT = (unsigned short*)(ws + 512 * 1024);    // 512 KB
    float*          hn   = (float*)(ws + 1024 * 1024);            // 4 KB

    vq_decompose_kernel<<<(K_DIM * D_DIM) / 256, 256, 0, stream>>>(emb, ehiT, eloT);
    vq_norms_kernel<<<K_DIM / 256, 256, 0, stream>>>(emb, hn, loss);
    vq_argmin_kernel<<<N_TOT / TILE_N, 256, 0, stream>>>(x, ehiT, eloT, hn, idxp);
    vq_gather_loss_kernel<<<32 * D_DIM, 256, 0, stream>>>(x, emb, idxp, out, loss);
}
